// RefineModule_68496138436770
// MI455X (gfx1250) — matrix operand order repaired and checked by reading, not compiled, not run on hardware
//
#include <hip/hip_runtime.h>

// ---------------- problem constants (fixed by the reference) ----------------
#define HH   128
#define WW   128
#define NBAT 4
#define CF   64      // feats channels
#define CM   21      // mask channels
#define NNB  24      // neighbors after remove_self (3 dilations x 8)
#define HW   (HH*WW)

// footprint for the 16-pixel row tile GEMM
#define KROWS 7               // dy in {-4,-2,-1,0,1,2,4}
#define KCOLS 24              // x0-4 .. x0+19
#define KTOT  (KROWS*KCOLS)   // 168
#define KPAD  192             // pad to 6 x K=32 WMMA chunks
#define BSTRT 200             // Bt row stride in halfs ([c][k] layout)
#define ASTR  200             // Als row stride in halfs ([m][k] layout)

typedef __attribute__((ext_vector_type(16))) _Float16 v16h;
typedef __attribute__((ext_vector_type(8)))  float    v8f;

// ---------------------------------------------------------------------------
// Phase 1: fused local_affinity (std + weighted |diff| mean + softmax).
// One block = one 16x16 pixel tile of one batch; loops all 64 channels with a
// 24x24 edge-clamped LDS tile (halo 4 covers dilation 4).
// ---------------------------------------------------------------------------
__global__ __launch_bounds__(256) void affinity_kernel(const float* __restrict__ feats,
                                                       float* __restrict__ aff) {
  const int bid = blockIdx.x;            // b*64 + ty*8 + tx
  const int tx = bid & 7;
  const int ty = (bid >> 3) & 7;
  const int b  = bid >> 6;
  const int tid = threadIdx.x;
  const int px = tid & 15, py = tid >> 4;
  const int gx = tx*16 + px, gy = ty*16 + py;

  __shared__ float tile[24*25];

  float acc[NNB];
#pragma unroll
  for (int n = 0; n < NNB; ++n) acc[n] = 0.f;

  for (int c = 0; c < CF; ++c) {
    __syncthreads();
    for (int i = tid; i < 24*24; i += 256) {
      int ly = i / 24, lx = i - ly*24;
      int sy = ty*16 + ly - 4; sy = sy < 0 ? 0 : (sy > HH-1 ? HH-1 : sy);
      int sx = tx*16 + lx - 4; sx = sx < 0 ? 0 : (sx > WW-1 ? WW-1 : sx);
      tile[ly*25 + lx] = feats[((b*CF + c)*HH + sy)*WW + sx];
    }
    __syncthreads();

    const int cy = py + 4, cx = px + 4;
    const float q = tile[cy*25 + cx];

    // pass 1: sample std over all 27 window values (centers included 3x)
    float s = 0.f, s2 = 0.f;
#pragma unroll
    for (int dd = 0; dd < 3; ++dd) {
      const int d = 1 << dd;
#pragma unroll
      for (int i2 = -1; i2 <= 1; ++i2)
#pragma unroll
        for (int j2 = -1; j2 <= 1; ++j2) {
          float v = tile[(cy + i2*d)*25 + (cx + j2*d)];
          s += v; s2 += v*v;
        }
    }
    float mean = s * (1.f/27.f);
    float var  = (s2 - s*mean) * (1.f/26.f);        // ddof=1
    var = fmaxf(var, 0.f);
    float invden = 1.f / (1e-8f + 0.1f * sqrtf(var));

    // pass 2: accumulate |k - q| * invden over the 24 non-self neighbors
    int n = 0;
#pragma unroll
    for (int dd = 0; dd < 3; ++dd) {
      const int d = 1 << dd;
#pragma unroll
      for (int t = 0; t < 9; ++t) {
        if (t == 4) continue;                       // remove self
        int i2 = t/3 - 1, j2 = (t - (t/3)*3) - 1;
        float v = tile[(cy + i2*d)*25 + (cx + j2*d)];
        acc[n] += fabsf(v - q) * invden;
        ++n;
      }
    }
  }

  // softmax over n of logits = -acc/64  (shift by max logit = -min(acc)/64)
  float mn = acc[0];
#pragma unroll
  for (int n = 1; n < NNB; ++n) mn = fminf(mn, acc[n]);
  float sum = 0.f;
#pragma unroll
  for (int n = 0; n < NNB; ++n) { acc[n] = __expf((mn - acc[n]) * (1.f/64.f)); sum += acc[n]; }
  float inv = 1.f / sum;
#pragma unroll
  for (int n = 0; n < NNB; ++n)
    aff[((b*NNB + n)*HH + gy)*WW + gx] = acc[n] * inv;
}

// ---------------------------------------------------------------------------
// Phase 2: one propagation iteration as a banded SpMM on the matrix cores.
// Block = (b, y, 16-wide x tile), 64 threads = 2 waves.
//   Bt[c][k] : mask values of the 7x24 clamped footprint (shared by the tile),
//              stored K-major so each lane's B fragment is two 8-half contiguous runs
//              (-> 2x ds_load_b128, like the A path)
//   Als[m][k]: pixel m's 24 affinity weights scattered (accumulated on clamp
//              collisions) into its footprint slots
//   D = A x B via 6 chunks of v_wmma_f32_16x16x32_f16; wave0 -> ch 0-15,
//   wave1 -> ch 16-20.
// ---------------------------------------------------------------------------
__global__ __launch_bounds__(64) void propagate_kernel(const float* __restrict__ src,
                                                       const float* __restrict__ aff,
                                                       float* __restrict__ dst) {
  const int bid = blockIdx.x;            // b*(128*8) + y*8 + xt
  const int xt = bid & 7;
  const int y  = (bid >> 3) & 127;
  const int b  = bid >> 10;
  const int x0 = xt * 16;
  const int tid = threadIdx.x;

  __shared__ _Float16 Bt[32 * BSTRT];    // [c][k], c 21..31 and k 168..191 = 0
  __shared__ _Float16 Als[16 * ASTR];    // [m][k], zero-initialized

  for (int i = tid; i < 32*BSTRT; i += 64) Bt[i]  = (_Float16)0.f;
  for (int i = tid; i < 16*ASTR;  i += 64) Als[i] = (_Float16)0.f;
  __syncthreads();

  const int dyv[KROWS] = {-4,-2,-1,0,1,2,4};

  // fill B: 168 footprint rows x 21 channels (k fastest => coalesced in x)
  for (int i = tid; i < KTOT*CM; i += 64) {
    int k = i % KTOT;
    int c = i / KTOT;
    int r = k / KCOLS, dxi = k - r*KCOLS;
    int yy = y + dyv[r];  yy = yy < 0 ? 0 : (yy > HH-1 ? HH-1 : yy);
    int xx = x0 + dxi - 4; xx = xx < 0 ? 0 : (xx > WW-1 ? WW-1 : xx);
    Bt[c*BSTRT + k] = (_Float16)src[((b*CM + c)*HH + yy)*WW + xx];
  }

  // fill A: one thread per pixel row so clamp-induced slot collisions
  // accumulate race-free (matches reference's duplicated clamped reads)
  if (tid < 16) {
    const int m = tid;
    const int x = x0 + m;
    const int di_[8] = {-1,-1,-1, 0, 0, 1, 1, 1};
    const int dj_[8] = {-1, 0, 1,-1, 1,-1, 0, 1};
    const int dil[3] = {1, 2, 4};
#pragma unroll
    for (int n = 0; n < NNB; ++n) {
      int d  = dil[n >> 3];
      int dy = di_[n & 7] * d;
      int r = 3;
#pragma unroll
      for (int q2 = 0; q2 < KROWS; ++q2) if (dyv[q2] == dy) r = q2;
      int xn = x + dj_[n & 7] * d; xn = xn < 0 ? 0 : (xn > WW-1 ? WW-1 : xn);
      int k = r*KCOLS + (xn - x0 + 4);
      float w = aff[((b*NNB + n)*HH + y)*WW + x];
      Als[m*ASTR + k] = (_Float16)((float)Als[m*ASTR + k] + w);
    }
  }
  __syncthreads();

  const int lane = tid & 31;
  const int wv   = tid >> 5;         // channel tile: 0 -> c0-15, 1 -> c16-20
  const int half = lane >> 4;
  const int lm   = lane & 15;
  const int cc   = wv*16 + lm;

  v8f acc = {};
#pragma unroll
  for (int chunk = 0; chunk < KPAD/32; ++chunk) {
    v16h af, bf;
    // A fragment (16-bit 16x32): M = lane%16; element j<8 = K base+j,
    // element 8+j = K base+16+j, with base = chunk*32 + 8*half. Both octets
    // are 8 contiguous-half, 16B-aligned LDS runs -> ds_load_b128 x2.
    const _Float16* ap = &Als[lm*ASTR + chunk*32 + 8*half];
    // B fragment (16-bit 32x16): N = lane%16 (channel cc); element j<8 = K
    // base+j, element 8+j = K base+16+j, with base = chunk*32 + 8*half.
    const _Float16* bp = &Bt[cc*BSTRT + chunk*32 + 8*half];
#pragma unroll
    for (int j = 0; j < 8; ++j) {
      af[j]     = ap[j];
      af[8 + j] = ap[16 + j];
    }
#pragma unroll
    for (int j = 0; j < 8; ++j) { bf[j] = bp[j]; bf[8 + j] = bp[16 + j]; }

    acc = __builtin_amdgcn_wmma_f32_16x16x32_f16(false, af, false, bf,
                                                 (short)0, acc, false, false);
  }

  // D layout: lane%16 = N (channel), VGPR v = pixel row v + 8*half
  if (cc < CM) {
#pragma unroll
    for (int v = 0; v < 8; ++v) {
      int m = v + 8*half;
      dst[((b*CM + cc)*HH + y)*WW + (x0 + m)] = acc[v];
    }
  }
}

// ---------------------------------------------------------------------------
extern "C" void kernel_launch(void* const* d_in, const int* in_sizes, int n_in,
                              void* d_out, int out_size, void* d_ws, size_t ws_size,
                              hipStream_t stream) {
  (void)in_sizes; (void)n_in; (void)out_size; (void)ws_size;
  const float* feats = (const float*)d_in[0];   // (4,64,128,128) f32
  const float* mask  = (const float*)d_in[1];   // (4,21,128,128) f32
  float* out  = (float*)d_out;                  // (4,21,128,128) f32
  float* affb = (float*)d_ws;                                   // 4*24*HW f32
  float* bufA = affb + (size_t)NBAT*NNB*HW;                     // 4*21*HW f32

  affinity_kernel<<<NBAT*8*8, 256, 0, stream>>>(feats, affb);

  // 10 iterations, ping-pong bufA <-> d_out; iter 9 (odd) lands in d_out
  const float* src = mask;
  for (int it = 0; it < 10; ++it) {
    float* dst = (it & 1) ? out : bufA;
    propagate_kernel<<<NBAT*HH*8, 64, 0, stream>>>(src, affb, dst);
    src = dst;
  }
}
